// InterpretableMultiHeadAttention_30485677867199
// MI455X (gfx1250) — matrix operand order repaired and checked by reading, not compiled, not run on hardware
//
#include <hip/hip_runtime.h>
#include <stdint.h>

// ---------------------------------------------------------------------------
// Multi-head attention forward for MI455X (gfx1250), wave32 + WMMA f16 + TDM.
//   x:(4,2048,1024) f32, mask:(4,1,2048,2048) bool, W_qkv:(2112,1024) f32,
//   W_out:(1024,1024) f32  ->  out:(4,2048,1024) f32
// ---------------------------------------------------------------------------

#define BATCH 4
#define SEQ   2048
#define HDIM  1024
#define NHEAD 16
#define DHEAD 64
#define QKVO  ((2 * NHEAD + 1) * DHEAD)   // 2112

typedef __attribute__((ext_vector_type(16))) _Float16 v16h;
typedef __attribute__((ext_vector_type(8)))  _Float16 v8h;
typedef __attribute__((ext_vector_type(4)))  _Float16 v4h;
typedef __attribute__((ext_vector_type(8)))  float    v8f;
typedef __attribute__((ext_vector_type(4)))  unsigned int v4u;
typedef __attribute__((ext_vector_type(8)))  int      v8i;
typedef __attribute__((ext_vector_type(4)))  int      v4i;

union AFrag { v16h v; v8h h[2]; };

static __device__ __forceinline__ v8f wmma16(v16h a, v16h b, v8f c) {
  // v_wmma_f32_16x16x32_f16: D = A(16x32) * B(32x16) + C(16x16 f32)
  return __builtin_amdgcn_wmma_f32_16x16x32_f16(
      /*neg_a=*/false, a, /*neg_b=*/false, b,
      /*c_mod=*/(short)0, c, /*reuse_a=*/false, /*reuse_b=*/false);
}

// --------------------------- TDM tile load ---------------------------------
// DMA a 64(contiguous dim0) x 64(rows) f16 tile of a row-major
// [tensor_rows x row_len] tensor into LDS at byte offset `lds_byte`, packed
// as LDS[row*64 + k].  D# layout per CDNA5 ISA ch.8: group0 = {count/flags,
// lds_addr, global_addr[31:0], global_addr[56:32] | type=2}; group1 packs
// data_size, tensor_dim0/1, tile_dim0/1/2, tensor_dim0_stride.
static __device__ __forceinline__ void tdm_load_tile64x64(
    const _Float16* gsrc, unsigned lds_byte, unsigned row_len,
    unsigned tensor_rows) {
  unsigned long long ga = (unsigned long long)(uintptr_t)gsrc;
  v4u g0 = { 1u,                                    // count=1, user mode
             lds_byte,                              // lds_addr (bytes)
             (unsigned)(ga & 0xFFFFFFFFu),
             (unsigned)((ga >> 32) & 0x1FFFFFFu) | (2u << 30) };  // type=2
  v8i g1 = { (int)(1u << 16),                       // data_size=1 -> 2 bytes
             (int)((row_len & 0xFFFFu) << 16),      // tensor_dim0[15:0]
             (int)((row_len >> 16) | ((tensor_rows & 0xFFFFu) << 16)),
             (int)((tensor_rows >> 16) | (64u << 16)),  // tile_dim0 = 64
             (int)(64u),                            // tile_dim1=64, tile_dim2=0
             (int)(row_len),                        // tensor_dim0_stride[31:0]
             0, 0 };
  v4i z4 = {0, 0, 0, 0};
#if defined(__clang_major__) && (__clang_major__ >= 23)
  v8i z8 = {0, 0, 0, 0, 0, 0, 0, 0};
  __builtin_amdgcn_tensor_load_to_lds(g0, g1, z4, z4, z8, 0);
#else
  __builtin_amdgcn_tensor_load_to_lds(g0, g1, z4, z4, 0);
#endif
}

// -------------------------------- fp32 -> f16 ------------------------------
__global__ void cvt_f32_f16(const float* __restrict__ in,
                            _Float16* __restrict__ out, int n) {
  int i = blockIdx.x * blockDim.x + threadIdx.x;
  int i4 = i * 4;
  if (i4 < n) {
    float4 f = *(const float4*)(in + i4);
    v4h h;
    h[0] = (_Float16)f.x; h[1] = (_Float16)f.y;
    h[2] = (_Float16)f.z; h[3] = (_Float16)f.w;
    *(v4h*)(out + i4) = h;
  }
}

// ------------------------- QKV projection GEMM -----------------------------
// qkv[m, o] = sum_k X[m,k] * W[o,k];  M=8192, N=2112, K=1024.
// Wave tile 64x64 (4x4 WMMA accum), 8 waves stacked along M -> WG tile 512x64.
// W tile (shared by all 8 waves) is staged through LDS with the Tensor Data
// Mover, double-buffered in 64-deep K slabs so the DMA overlaps the MMAs.
// Scatter-store: q,k -> (b, head, t, d) head-major; v -> (b, t, d).
__global__ __launch_bounds__(256) void qkv_gemm(
    const _Float16* __restrict__ X,   // 8192 x 1024
    const _Float16* __restrict__ W,   // 2112 x 1024
    _Float16* __restrict__ Q,         // (B*NHEAD, SEQ, DHEAD)
    _Float16* __restrict__ Kh,        // (B*NHEAD, SEQ, DHEAD)
    _Float16* __restrict__ V)         // (B, SEQ, DHEAD)
{
  __shared__ __attribute__((aligned(128))) _Float16 Bl[2][64 * 64]; // 2x8 KB

  const int lane = threadIdx.x & 31;
  const int wave = threadIdx.x >> 5;
  const int lm = lane & 15;          // M/N position inside 16-wide tile
  const int lh = lane >> 4;          // half-wave select
  const int row0 = blockIdx.y * 512 + wave * 64;
  const int col0 = blockIdx.x * 64;

  // ISA 10.2: flat LDS addresses map to LDS by truncation to [31:0].
  const unsigned lds0 = (unsigned)(uintptr_t)&Bl[0][0];
  const unsigned lds1 = (unsigned)(uintptr_t)&Bl[1][0];

  v8f acc[4][4];
#pragma unroll
  for (int i = 0; i < 4; ++i)
#pragma unroll
    for (int j = 0; j < 4; ++j) acc[i][j] = (v8f)0.0f;

  if (wave == 0)
    tdm_load_tile64x64(W + (size_t)col0 * HDIM, lds0, HDIM, QKVO);
  __builtin_amdgcn_s_wait_tensorcnt(0);
  __syncthreads();

  for (int s = 0; s < HDIM / 64; ++s) {
    const int kslab = s * 64;
    const _Float16* Bbuf = Bl[s & 1];
    if (s + 1 < HDIM / 64 && wave == 0) {    // prefetch next slab via TDM
      unsigned ldsNext = ((s + 1) & 1) ? lds1 : lds0;
      tdm_load_tile64x64(W + (size_t)col0 * HDIM + kslab + 64, ldsNext,
                         HDIM, QKVO);
    }
#pragma unroll
    for (int kk = 0; kk < 2; ++kk) {
      const int k0 = kslab + kk * 32;
      // A fragments: contiguous k order (same as B) -> lane holds K = k0+lh*16 .. +15
      AFrag a[4];
#pragma unroll
      for (int i = 0; i < 4; ++i) {
        const _Float16* p =
            X + (size_t)(row0 + i * 16 + lm) * HDIM + k0 + lh * 16;
        a[i].h[0] = *(const v8h*)p;
        a[i].h[1] = *(const v8h*)(p + 8);
      }
      // B fragments from LDS: lane holds col (j*16+lm), K = k0+lh*16 .. +15
      v16h b[4];
#pragma unroll
      for (int j = 0; j < 4; ++j)
        b[j] = *(const v16h*)&Bbuf[(j * 16 + lm) * 64 + kk * 32 + lh * 16];
      if (k0 + 32 < HDIM)  // global_prefetch_b8, near-cache hint
        __builtin_prefetch(X + (size_t)(row0 + lm) * HDIM + k0 + 32, 0, 3);
#pragma unroll
      for (int i = 0; i < 4; ++i)
#pragma unroll
        for (int j = 0; j < 4; ++j)
          acc[i][j] = wmma16(a[i].v, b[j], acc[i][j]);
    }
    if (s + 1 < HDIM / 64) {
      __builtin_amdgcn_s_wait_tensorcnt(0);  // waves 1..7 pass trivially
      __syncthreads();                       // publish DMA'd slab to all waves
    }
  }

  // C layout: element e of VGPR group -> row (lh*8 + e), col lm.
#pragma unroll
  for (int i = 0; i < 4; ++i) {
#pragma unroll
    for (int j = 0; j < 4; ++j) {
#pragma unroll
      for (int e = 0; e < 8; ++e) {
        int grow = row0 + i * 16 + lh * 8 + e;     // global token index
        int gcol = col0 + j * 16 + lm;             // output feature
        int bb = grow >> 11, t = grow & (SEQ - 1);
        _Float16 hv = (_Float16)acc[i][j][e];
        if (gcol < NHEAD * DHEAD) {
          int head = gcol >> 6, dd = gcol & 63;
          Q[(((size_t)bb * NHEAD + head) * SEQ + t) * DHEAD + dd] = hv;
        } else if (gcol < 2 * NHEAD * DHEAD) {
          int oc = gcol - NHEAD * DHEAD;
          int head = oc >> 6, dd = oc & 63;
          Kh[(((size_t)bb * NHEAD + head) * SEQ + t) * DHEAD + dd] = hv;
        } else {
          V[((size_t)bb * SEQ + t) * DHEAD + (gcol - 2 * NHEAD * DHEAD)] = hv;
        }
      }
    }
  }
}

// ----------------------- fused flash attention -----------------------------
// grid.x = b*NHEAD, grid.y = q tile of 128 rows (8 waves x 16 q-rows each).
// Online softmax in fp32; S and O stay in WMMA C-layout; P bounced through a
// per-wave LDS slab (C->A relayout).  The 64x64 K chunk (identical for all 8
// waves) rides a double-buffered TDM pipeline; the V chunk is cooperatively
// staged transposed so every V B-fragment is one contiguous LDS read.
__global__ __launch_bounds__(256) void flash_attn(
    const _Float16* __restrict__ Q,
    const _Float16* __restrict__ Kh,
    const _Float16* __restrict__ V,
    const unsigned char* __restrict__ mask,   // (B, SEQ, SEQ) bool
    _Float16* __restrict__ Attn)              // (B*SEQ, HDIM)
{
  __shared__ __attribute__((aligned(128))) _Float16 Kl[2][64 * 64]; // 16 KB
  __shared__ __attribute__((aligned(32)))  _Float16 Pl[8][16 * 64]; // 16 KB
  __shared__ __attribute__((aligned(32)))  _Float16 Vt[64 * 64];    //  8 KB

  const int lane = threadIdx.x & 31;
  const int wave = threadIdx.x >> 5;
  const int lm = lane & 15;
  const int lh = lane >> 4;
  const int bh = blockIdx.x;
  const int b = bh >> 4, head = bh & 15;
  const int q0 = blockIdx.y * 128 + wave * 16;
  const float scale = 0.125f;                 // 1/sqrt(64)
  const float NEGMAX = -3.4028235e38f;        // matches nan_to_num(-inf)

  const _Float16* qptr = Q + (size_t)bh * SEQ * DHEAD;
  const _Float16* kptr = Kh + (size_t)bh * SEQ * DHEAD;
  const _Float16* vptr = V + (size_t)b * SEQ * DHEAD;
  const unsigned char* mptr = mask + (size_t)b * SEQ * SEQ;

  const unsigned ldsK0 = (unsigned)(uintptr_t)&Kl[0][0];
  const unsigned ldsK1 = (unsigned)(uintptr_t)&Kl[1][0];

  // Q A-fragments for d-chunks {0,32}; resident for the whole kv sweep.
  AFrag qa[2];
#pragma unroll
  for (int c = 0; c < 2; ++c) {
    const _Float16* p = qptr + (size_t)(q0 + lm) * DHEAD + c * 32 + lh * 16;
    qa[c].h[0] = *(const v8h*)p;
    qa[c].h[1] = *(const v8h*)(p + 8);
  }

  float mrun[8], lrun[8];
  v8f o[4];
#pragma unroll
  for (int e = 0; e < 8; ++e) { mrun[e] = NEGMAX; lrun[e] = 0.0f; }
#pragma unroll
  for (int n = 0; n < 4; ++n) o[n] = (v8f)0.0f;

  if (wave == 0)   // prologue: DMA first K chunk
    tdm_load_tile64x64(kptr, ldsK0, DHEAD, SEQ);

  const int NCH = SEQ / 64;   // 32 kv chunks
  for (int s = 0; s < NCH; ++s) {
    const int kv0 = s * 64;
    const _Float16* Kbuf = Kl[s & 1];

    // ---- cooperative V chunk stage: Vt[d*64 + kv] (transposed) ----
    __syncthreads();                       // prior chunk fully consumed
    {
      int kvr = threadIdx.x >> 2;          // 0..63
      int d0 = (threadIdx.x & 3) * 16;     // 0,16,32,48
      v16h row = *(const v16h*)(vptr + (size_t)(kv0 + kvr) * DHEAD + d0);
#pragma unroll
      for (int i = 0; i < 16; ++i) Vt[(d0 + i) * 64 + kvr] = row[i];
    }
    // ---- K pipeline: issue chunk s+1, retire chunk s (in-order TDM) ----
    if (s + 1 < NCH) {
      if (wave == 0)
        tdm_load_tile64x64(kptr + (size_t)(kv0 + 64) * DHEAD,
                           ((s + 1) & 1) ? ldsK1 : ldsK0, DHEAD, SEQ);
      __builtin_amdgcn_s_wait_tensorcnt(1);   // chunk s retired
    } else {
      __builtin_amdgcn_s_wait_tensorcnt(0);
    }
    __syncthreads();                          // publish Kl[s&1] + Vt

    // ---- S = Q * K^T over this 16 x 64 chunk (K from LDS) ----
    v8f s4[4];
#pragma unroll
    for (int j = 0; j < 4; ++j) s4[j] = (v8f)0.0f;
#pragma unroll
    for (int j = 0; j < 4; ++j) {
#pragma unroll
      for (int c = 0; c < 2; ++c) {
        v16h kf = *(const v16h*)&Kbuf[(j * 16 + lm) * 64 + c * 32 + lh * 16];
        s4[j] = wmma16(qa[c].v, kf, s4[j]);
      }
    }
    // ---- scale + mask ----
#pragma unroll
    for (int j = 0; j < 4; ++j) {
#pragma unroll
      for (int e = 0; e < 8; ++e) {
        int qr = q0 + lh * 8 + e;
        int kc = kv0 + j * 16 + lm;
        float sv = s4[j][e] * scale;
        if (mptr[(size_t)qr * SEQ + kc]) sv = NEGMAX;
        s4[j][e] = sv;
      }
    }
    // ---- online softmax: row stats replicated across each 16-lane half ----
    float mnew[8];
#pragma unroll
    for (int e = 0; e < 8; ++e) {
      float v = fmaxf(fmaxf(s4[0][e], s4[1][e]), fmaxf(s4[2][e], s4[3][e]));
      v = fmaxf(v, __shfl_xor(v, 1, 16));
      v = fmaxf(v, __shfl_xor(v, 2, 16));
      v = fmaxf(v, __shfl_xor(v, 4, 16));
      v = fmaxf(v, __shfl_xor(v, 8, 16));
      mnew[e] = fmaxf(mrun[e], v);
    }
#pragma unroll
    for (int j = 0; j < 4; ++j)
#pragma unroll
      for (int e = 0; e < 8; ++e) s4[j][e] = __expf(s4[j][e] - mnew[e]);
#pragma unroll
    for (int e = 0; e < 8; ++e) {
      float r = s4[0][e] + s4[1][e] + s4[2][e] + s4[3][e];
      r += __shfl_xor(r, 1, 16);
      r += __shfl_xor(r, 2, 16);
      r += __shfl_xor(r, 4, 16);
      r += __shfl_xor(r, 8, 16);
      float alpha = __expf(mrun[e] - mnew[e]);
      lrun[e] = lrun[e] * alpha + r;
      mrun[e] = mnew[e];
#pragma unroll
      for (int n = 0; n < 4; ++n) o[n][e] *= alpha;   // rescale O rows
    }
    // ---- P: C-layout -> per-wave LDS -> A-layout ----
#pragma unroll
    for (int j = 0; j < 4; ++j)
#pragma unroll
      for (int e = 0; e < 8; ++e)
        Pl[wave][(lh * 8 + e) * 64 + j * 16 + lm] = (_Float16)s4[j][e];
    asm volatile("" ::: "memory");   // same-wave DS ops are in-order
    // ---- O += P * V ----
#pragma unroll
    for (int c = 0; c < 2; ++c) {
      AFrag pa;
      const _Float16* pp = &Pl[wave][lm * 64 + c * 32 + lh * 16];
      pa.h[0] = *(const v8h*)pp;
      pa.h[1] = *(const v8h*)(pp + 8);
#pragma unroll
      for (int n = 0; n < 4; ++n) {
        v16h vf = *(const v16h*)&Vt[(n * 16 + lm) * 64 + c * 32 + lh * 16];
        o[n] = wmma16(pa.v, vf, o[n]);
      }
    }
  }

  // ---- normalize + store to (b, t, head*64 + d) f16 ----
#pragma unroll
  for (int n = 0; n < 4; ++n) {
#pragma unroll
    for (int e = 0; e < 8; ++e) {
      int qr = q0 + lh * 8 + e;
      float val = o[n][e] / lrun[e];
      Attn[((size_t)(b * SEQ) + qr) * HDIM + head * DHEAD + n * 16 + lm] =
          (_Float16)val;
    }
  }
}

// ------------------------- output projection GEMM --------------------------
// out[m, o] = sum_k attn[m,k] * W_out[o,k];  M=8192, N=1024, K=1024, f32 out.
// Same TDM-staged, double-buffered weight pipeline as qkv_gemm.
__global__ __launch_bounds__(256) void out_gemm(
    const _Float16* __restrict__ A,   // 8192 x 1024
    const _Float16* __restrict__ W,   // 1024 x 1024
    float* __restrict__ Out)          // 8192 x 1024
{
  __shared__ __attribute__((aligned(128))) _Float16 Bl[2][64 * 64]; // 2x8 KB

  const int lane = threadIdx.x & 31;
  const int wave = threadIdx.x >> 5;
  const int lm = lane & 15;
  const int lh = lane >> 4;
  const int row0 = blockIdx.y * 512 + wave * 64;
  const int col0 = blockIdx.x * 64;

  const unsigned lds0 = (unsigned)(uintptr_t)&Bl[0][0];
  const unsigned lds1 = (unsigned)(uintptr_t)&Bl[1][0];

  v8f acc[4][4];
#pragma unroll
  for (int i = 0; i < 4; ++i)
#pragma unroll
    for (int j = 0; j < 4; ++j) acc[i][j] = (v8f)0.0f;

  if (wave == 0)
    tdm_load_tile64x64(W + (size_t)col0 * HDIM, lds0, HDIM, HDIM);
  __builtin_amdgcn_s_wait_tensorcnt(0);
  __syncthreads();

  for (int s = 0; s < HDIM / 64; ++s) {
    const int kslab = s * 64;
    const _Float16* Bbuf = Bl[s & 1];
    if (s + 1 < HDIM / 64 && wave == 0) {
      unsigned ldsNext = ((s + 1) & 1) ? lds1 : lds0;
      tdm_load_tile64x64(W + (size_t)col0 * HDIM + kslab + 64, ldsNext,
                         HDIM, HDIM);
    }
#pragma unroll
    for (int kk = 0; kk < 2; ++kk) {
      const int k0 = kslab + kk * 32;
      AFrag a[4];
#pragma unroll
      for (int i = 0; i < 4; ++i) {
        const _Float16* p =
            A + (size_t)(row0 + i * 16 + lm) * HDIM + k0 + lh * 16;
        a[i].h[0] = *(const v8h*)p;
        a[i].h[1] = *(const v8h*)(p + 8);
      }
      v16h b[4];
#pragma unroll
      for (int j = 0; j < 4; ++j)
        b[j] = *(const v16h*)&Bbuf[(j * 16 + lm) * 64 + kk * 32 + lh * 16];
      if (k0 + 32 < HDIM)
        __builtin_prefetch(A + (size_t)(row0 + lm) * HDIM + k0 + 32, 0, 3);
#pragma unroll
      for (int i = 0; i < 4; ++i)
#pragma unroll
        for (int j = 0; j < 4; ++j)
          acc[i][j] = wmma16(a[i].v, b[j], acc[i][j]);
    }
    if (s + 1 < HDIM / 64) {
      __builtin_amdgcn_s_wait_tensorcnt(0);
      __syncthreads();
    }
  }

#pragma unroll
  for (int i = 0; i < 4; ++i)
#pragma unroll
    for (int j = 0; j < 4; ++j)
#pragma unroll
      for (int e = 0; e < 8; ++e) {
        int grow = row0 + i * 16 + lh * 8 + e;
        int gcol = col0 + j * 16 + lm;
        Out[(size_t)grow * HDIM + gcol] = acc[i][j][e];
      }
}

// ---------------------------------------------------------------------------
extern "C" void kernel_launch(void* const* d_in, const int* in_sizes, int n_in,
                              void* d_out, int out_size, void* d_ws,
                              size_t ws_size, hipStream_t stream) {
  const float* x = (const float*)d_in[0];                       // (4,2048,1024)
  const unsigned char* mask = (const unsigned char*)d_in[1];    // bool (4,1,T,T)
  const float* wqkv = (const float*)d_in[2];                    // (2112,1024)
  const float* wout = (const float*)d_in[3];                    // (1024,1024)
  float* out = (float*)d_out;

  const size_t M = (size_t)BATCH * SEQ;                         // 8192
  const size_t qkElems = (size_t)BATCH * NHEAD * SEQ * DHEAD;   // 8.4M

  char* ws = (char*)d_ws;
  size_t off = 0;
  _Float16* xh    = (_Float16*)(ws + off); off += M * HDIM * 2;              // 16 MB
  _Float16* wqkvh = (_Float16*)(ws + off); off += (size_t)QKVO * HDIM * 2;   // 4.3 MB
  _Float16* wouth = (_Float16*)(ws + off); off += (size_t)HDIM * HDIM * 2;   // 2 MB
  _Float16* qh    = (_Float16*)(ws + off); off += qkElems * 2;               // 16 MB
  _Float16* kh    = (_Float16*)(ws + off); off += qkElems * 2;               // 16 MB
  _Float16* vh    = (_Float16*)(ws + off); off += (size_t)BATCH * SEQ * DHEAD * 2;
  _Float16* attnh = (_Float16*)(ws + off); off += M * HDIM * 2;              // 16 MB
  // total ~71 MB of workspace

  // 1) downconvert inputs to f16 (4 elements / thread)
  {
    int n = (int)(M * HDIM);
    cvt_f32_f16<<<(n / 4 + 255) / 256, 256, 0, stream>>>(x, xh, n);
    n = QKVO * HDIM;
    cvt_f32_f16<<<(n / 4 + 255) / 256, 256, 0, stream>>>(wqkv, wqkvh, n);
    n = HDIM * HDIM;
    cvt_f32_f16<<<(n / 4 + 255) / 256, 256, 0, stream>>>(wout, wouth, n);
  }
  // 2) QKV projection: grid = (2112/64, 8192/512), TDM-staged weights
  qkv_gemm<<<dim3(QKVO / 64, M / 512), 256, 0, stream>>>(xh, wqkvh, qh, kh, vh);
  // 3) fused attention: grid = (B*NHEAD, SEQ/128), TDM-staged K chunks
  flash_attn<<<dim3(BATCH * NHEAD, SEQ / 128), 256, 0, stream>>>(qh, kh, vh,
                                                                 mask, attnh);
  // 4) output projection -> fp32 d_out
  out_gemm<<<dim3(HDIM / 64, M / 512), 256, 0, stream>>>(attnh, wouth, out);
}
